// Block_87179246174512
// MI455X (gfx1250) — matrix operand order repaired and checked by reading, not compiled, not run on hardware
//
#include <hip/hip_runtime.h>
#include <math.h>

typedef _Float16 half_t;
typedef __attribute__((ext_vector_type(16))) _Float16 v16h;
typedef __attribute__((ext_vector_type(8)))  _Float16 v8h;
typedef __attribute__((ext_vector_type(8)))  float    v8f;
typedef __attribute__((ext_vector_type(4)))  unsigned int u32x4;
typedef __attribute__((ext_vector_type(8)))  int          i32x8;
typedef __attribute__((ext_vector_type(4)))  int          i32x4;

#define WMMA_F16(a, b, c) \
  __builtin_amdgcn_wmma_f32_16x16x32_f16(false, (a), false, (b), (short)0, (c), false, false)

#if __has_builtin(__builtin_amdgcn_tensor_load_to_lds) && \
    __has_builtin(__builtin_amdgcn_s_wait_tensorcnt)
#define HAVE_TDM 1
#else
#define HAVE_TDM 0
#endif

static constexpr int NB   = 8;      // batch
static constexpr int T    = 1024;   // seq len
static constexpr int C    = 768;    // embed
static constexpr int H    = 12;     // heads
static constexpr int HD   = 64;     // head dim
static constexpr int C3   = 3 * C;  // 2304
static constexpr int MTOK = NB * T; // 8192 tokens

__device__ __forceinline__ float gelu_tanh(float x) {
  float x3 = x * x * x;
  return 0.5f * x * (1.0f + tanhf(0.7978845608028654f * (x + 0.044715f * x3)));
}

__device__ __forceinline__ v16h cat8(v8h a, v8h b) {
  return __builtin_shufflevector(a, b, 0, 1, 2, 3, 4, 5, 6, 7, 8, 9, 10, 11, 12, 13, 14, 15);
}

// A-fragment (16x32 f16): lane -> row; elems 0..7: K=hi*8+e, elems 8..15: K=16+hi*8+e.
__device__ __forceinline__ v16h load_afrag(const half_t* rowp, int hi) {
  v8h lo = *(const v8h*)(rowp + hi * 8);
  v8h hh = *(const v8h*)(rowp + 16 + hi * 8);
  return cat8(lo, hh);
}

// B-fragment (32x16 f16) from a TRANSPOSED tile T[N][K]: lane -> col n; elems 0..7: K=hi*8+e, elems 8..15: K=16+hi*8+e.
__device__ __forceinline__ v16h load_bfrag(const half_t* colp, int hi) {
  v8h lo = *(const v8h*)(colp + hi * 8);
  v8h hh = *(const v8h*)(colp + 16 + hi * 8);
  return cat8(lo, hh);
}

// ---------------------------------------------------------------------------
// TDM: async 2-D tile load (f16) global -> LDS. tile = 64 x tile_d1 elements,
// padded 4 dwords per 32 dwords => LDS row pitch 72 f16.
// ---------------------------------------------------------------------------
__device__ __forceinline__ void tdm_load_tile(const half_t* gaddr, unsigned lds_off,
                                              unsigned tensor_d0, unsigned tensor_d1,
                                              unsigned stride_elems, unsigned tile_d1) {
#if HAVE_TDM
  unsigned long long ga = (unsigned long long)(size_t)gaddr;
  u32x4 g0;
  g0[0] = 1u;                                   // count=1, user mode
  g0[1] = lds_off;                              // lds_addr (bytes)
  g0[2] = (unsigned)(ga & 0xFFFFFFFFu);         // global_addr[31:0]
  g0[3] = (unsigned)((ga >> 32) & 0x1FFFFFFu)   // global_addr[56:32]
          | (2u << 30);                         // type=2 (image)
  i32x8 g1;
  g1[0] = (int)((1u << 16)      // data_size = 1 -> 2 bytes
                | (1u << 20)    // pad_enable
                | (4u << 22)    // pad_interval code 4 -> 32 dwords (one 64-f16 row)
                | (3u << 25));  // pad_amount  code 3 -> 4 dwords (8 f16)
  g1[1] = (int)((tensor_d0 & 0xFFFFu) << 16);                                   // dim0[15:0]
  g1[2] = (int)(((tensor_d0 >> 16) & 0xFFFFu) | ((tensor_d1 & 0xFFFFu) << 16)); // dim0[31:16]|dim1[15:0]
  g1[3] = (int)(((tensor_d1 >> 16) & 0xFFFFu) | (64u << 16));                   // dim1[31:16]|tile_dim0=64
  g1[4] = (int)(tile_d1 & 0xFFFFu);             // tile_dim1, tile_dim2=0
  g1[5] = (int)stride_elems;                    // tensor_dim0_stride[31:0]
  g1[6] = 0;                                    // stride[47:32]=0, dim1_stride[15:0]=0
  g1[7] = 0;
  i32x4 z4 = {0, 0, 0, 0};
  i32x8 z8 = {0, 0, 0, 0, 0, 0, 0, 0};
  __builtin_amdgcn_tensor_load_to_lds(g0, g1, z4, z4, z8, 0);
#else
  (void)gaddr; (void)lds_off; (void)tensor_d0; (void)tensor_d1; (void)stride_elems; (void)tile_d1;
#endif
}

// ---------------------------------------------------------------------------
// fp32 [K][N] -> f16 transposed [N][K] (for GEMM B operands)
// ---------------------------------------------------------------------------
__global__ __launch_bounds__(256) void cvt_t_f16_k(const float* __restrict__ in,
                                                   half_t* __restrict__ out, int K, int N) {
  __shared__ float tile[32][33];
  const int n0 = blockIdx.x * 32, k0 = blockIdx.y * 32;
  const int tx = threadIdx.x & 31, ty = threadIdx.x >> 5;
#pragma unroll
  for (int i = 0; i < 32; i += 8)
    tile[ty + i][tx] = in[(size_t)(k0 + ty + i) * N + n0 + tx];
  __syncthreads();
#pragma unroll
  for (int i = 0; i < 32; i += 8)
    out[(size_t)(n0 + ty + i) * K + k0 + tx] = (half_t)tile[tx][ty + i];
}

// ---------------------------------------------------------------------------
// LayerNorm over last dim (768) -> f16. One row per block, 256 threads.
// ---------------------------------------------------------------------------
__global__ __launch_bounds__(256) void ln_f16_k(const float* __restrict__ x,
                                                const float* __restrict__ g,
                                                const float* __restrict__ bta,
                                                half_t* __restrict__ out) {
  int row = blockIdx.x;
  int t = threadIdx.x;
  const float* xr = x + (size_t)row * C;
  float v0 = xr[t], v1 = xr[t + 256], v2 = xr[t + 512];
  __shared__ float rs[256], rq[256];
  rs[t] = v0 + v1 + v2;
  rq[t] = v0 * v0 + v1 * v1 + v2 * v2;
  __syncthreads();
  for (int st = 128; st > 0; st >>= 1) {
    if (t < st) { rs[t] += rs[t + st]; rq[t] += rq[t + st]; }
    __syncthreads();
  }
  float mean = rs[0] * (1.0f / C);
  float var  = rq[0] * (1.0f / C) - mean * mean;
  float rstd = rsqrtf(var + 1e-5f);
  half_t* orow = out + (size_t)row * C;
  orow[t]       = (half_t)((v0 - mean) * rstd * g[t]       + bta[t]);
  orow[t + 256] = (half_t)((v1 - mean) * rstd * g[t + 256] + bta[t + 256]);
  orow[t + 512] = (half_t)((v2 - mean) * rstd * g[t + 512] + bta[t + 512]);
}

// ---------------------------------------------------------------------------
// WMMA GEMM: Cout[M,N] = A[M,K](f16) x Bt[N,K](f16, transposed) + bias,
// optional GELU, optional residual add (f32), f16 or f32 output.
// Block = 256 threads = 8 waves; workgroup tile 128x64 (MxN), BK=64.
// Wave (wrow,wcol) owns a 32x32 C patch -> 4 accumulators; A/B frags reused 2x.
// TDM double-buffered tile loads.
// ---------------------------------------------------------------------------
template <bool OUT_F16, bool GELU, bool RESID>
__global__ __launch_bounds__(256) void gemm_wmma_k(const half_t* __restrict__ A,
                                                   const half_t* __restrict__ Bt,
                                                   const float* __restrict__ bias,
                                                   const float* __restrict__ resid,
                                                   void* __restrict__ Cout,
                                                   int M, int N, int K) {
  __shared__ __align__(16) half_t As[2][128][72];
  __shared__ __align__(16) half_t Bs[2][64][72];

  const int tid = threadIdx.x;
  const int m0 = blockIdx.y * 128;
  const int n0 = blockIdx.x * 64;

  const int w    = tid >> 5;
  const int lane = tid & 31;
  const int ml   = lane & 15;
  const int hi   = lane >> 4;
  const int wrow = w >> 1;  // 0..3 -> 32-row band
  const int wcol = w & 1;   // 0..1 -> 32-col band

  // opaque never-true store: keeps LDS (written only by TDM) from being folded away
  if ((int)blockDim.y == 7777) { As[0][0][0] = (half_t)1; Bs[0][0][0] = (half_t)1; }

  v8f acc[2][2];
#pragma unroll
  for (int mt = 0; mt < 2; mt++)
#pragma unroll
    for (int nt = 0; nt < 2; nt++) {
      v8f z = {0.f, 0.f, 0.f, 0.f, 0.f, 0.f, 0.f, 0.f};
      acc[mt][nt] = z;
    }

  const int niter = K >> 6;

#if HAVE_TDM
  if (w == 0) {
    tdm_load_tile(&A [(size_t)m0 * K], (unsigned)(size_t)&As[0][0][0], (unsigned)K, 128u,
                  (unsigned)K, 128u);
    tdm_load_tile(&Bt[(size_t)n0 * K], (unsigned)(size_t)&Bs[0][0][0], (unsigned)K, 64u,
                  (unsigned)K, 64u);
  }
#endif

  for (int i = 0; i < niter; i++) {
    const int bi = i & 1;
#if HAVE_TDM
    if (w == 0) {
      if (i + 1 < niter) {
        const int k1 = (i + 1) << 6;
        tdm_load_tile(&A [(size_t)m0 * K + k1], (unsigned)(size_t)&As[bi ^ 1][0][0],
                      (unsigned)K, 128u, (unsigned)K, 128u);
        tdm_load_tile(&Bt[(size_t)n0 * K + k1], (unsigned)(size_t)&Bs[bi ^ 1][0][0],
                      (unsigned)K, 64u, (unsigned)K, 64u);
        __builtin_amdgcn_s_wait_tensorcnt(2);  // oldest tile (i) complete
      } else {
        __builtin_amdgcn_s_wait_tensorcnt(0);
      }
    }
    __syncthreads();  // tile i visible to all waves
#else
    const int k0 = i << 6;
    __syncthreads();
    {
      const int arr = tid >> 1, acc2 = (tid & 1) * 32;  // A: 128 rows x 64
#pragma unroll
      for (int c4 = 0; c4 < 32; c4 += 8)
        *(v8h*)&As[bi][arr][acc2 + c4] = *(const v8h*)&A[(size_t)(m0 + arr) * K + k0 + acc2 + c4];
      const int brr = tid >> 2, bcc = (tid & 3) * 16;   // B: 64 rows x 64
      *(v8h*)&Bs[bi][brr][bcc]     = *(const v8h*)&Bt[(size_t)(n0 + brr) * K + k0 + bcc];
      *(v8h*)&Bs[bi][brr][bcc + 8] = *(const v8h*)&Bt[(size_t)(n0 + brr) * K + k0 + bcc + 8];
    }
    __syncthreads();
#endif

    const half_t(*At)[72]  = As[bi];
    const half_t(*Bts)[72] = Bs[bi];
#pragma unroll
    for (int kc = 0; kc < 64; kc += 32) {
      v16h afr[2];
#pragma unroll
      for (int mt = 0; mt < 2; mt++)
        afr[mt] = load_afrag(&At[wrow * 32 + mt * 16 + ml][kc], hi);
#pragma unroll
      for (int nt = 0; nt < 2; nt++) {
        v16h bfr = load_bfrag(&Bts[wcol * 32 + nt * 16 + ml][kc], hi);
#pragma unroll
        for (int mt = 0; mt < 2; mt++)
          acc[mt][nt] = WMMA_F16(afr[mt], bfr, acc[mt][nt]);
      }
    }
    __syncthreads();  // done with buffer bi before it is refilled
  }

  // C layout: lane -> col N=ml; VGPR r -> row M = hi*8 + r
#pragma unroll
  for (int mt = 0; mt < 2; mt++) {
#pragma unroll
    for (int nt = 0; nt < 2; nt++) {
#pragma unroll
      for (int r = 0; r < 8; r++) {
        int row = m0 + wrow * 32 + mt * 16 + hi * 8 + r;
        int col = n0 + wcol * 32 + nt * 16 + ml;
        size_t idx = (size_t)row * N + col;
        float v = acc[mt][nt][r] + bias[col];
        if (GELU)  v = gelu_tanh(v);
        if (RESID) v += resid[idx];
        if (OUT_F16) ((half_t*)Cout)[idx] = (half_t)v;
        else         ((float*)Cout)[idx] = v;
      }
    }
  }
}

// ---------------------------------------------------------------------------
// Flash attention (causal, UNSCALED scores per reference).
// Grid: (T/128, NB*H). Block 256 = 8 waves; each wave owns 16 query rows.
// ---------------------------------------------------------------------------
__global__ __launch_bounds__(256) void attn_wmma_k(const half_t* __restrict__ qkv,
                                                   half_t* __restrict__ outp) {
  __shared__ __align__(16) half_t Qs[128][72];
  __shared__ __align__(16) half_t Ks[64][72];     // [key][hd]
  __shared__ __align__(16) half_t VT[64][72];     // [hd][key]
  __shared__ __align__(16) half_t Ps[8][16][72];  // per-wave P staging [qrow][key]

  const int tid = threadIdx.x;
  const int qb  = blockIdx.x;
  const int bh  = blockIdx.y;
  const int b   = bh / H;
  const int h   = bh % H;
  const int q0  = qb * 128;

  const half_t* Qg = qkv + (size_t)b * T * C3 + h * HD;
  const half_t* Kg = Qg + C;
  const half_t* Vg = Qg + 2 * C;

  const int w    = tid >> 5;
  const int lane = tid & 31;
  const int ml   = lane & 15;
  const int hi   = lane >> 4;
  const int qw   = q0 + w * 16;

  half_t(*Pw)[72] = Ps[w];

  {
    const int c = (tid & 7) * 8;
#pragma unroll
    for (int i = 0; i < 4; i++) {
      int r = (tid >> 3) + 32 * i;
      *(v8h*)&Qs[r][c] = *(const v8h*)&Qg[(size_t)(q0 + r) * C3 + c];
    }
  }

  v8f o[4];
#pragma unroll
  for (int i = 0; i < 4; i++) {
    v8f z = {0.f, 0.f, 0.f, 0.f, 0.f, 0.f, 0.f, 0.f};
    o[i] = z;
  }
  float m_run[8], l_run[8];
#pragma unroll
  for (int r = 0; r < 8; r++) { m_run[r] = -__builtin_inff(); l_run[r] = 0.0f; }

  const int jend = q0 + 128;
  for (int j0 = 0; j0 < jend; j0 += 64) {
    __syncthreads();
    {
      const int c = (tid & 7) * 8;
#pragma unroll
      for (int i2 = 0; i2 < 2; i2++) {
        int r = (tid >> 3) + 32 * i2;
        *(v8h*)&Ks[r][c] = *(const v8h*)&Kg[(size_t)(j0 + r) * C3 + c];
        v8h vv = *(const v8h*)&Vg[(size_t)(j0 + r) * C3 + c];
#pragma unroll
        for (int e = 0; e < 8; e++) VT[c + e][r] = vv[e];
        if (j0 + 64 < jend) {
          __builtin_prefetch(&Kg[(size_t)(j0 + 64 + r) * C3 + c], 0, 0);
          __builtin_prefetch(&Vg[(size_t)(j0 + 64 + r) * C3 + c], 0, 0);
        }
      }
    }
    __syncthreads();

    // ---- S = Q x K^T
    v16h qa0 = load_afrag(&Qs[w * 16 + ml][0], hi);
    v16h qa1 = load_afrag(&Qs[w * 16 + ml][32], hi);
    v8f sacc[4];
#pragma unroll
    for (int nt = 0; nt < 4; nt++) {
      v8f s = {0.f, 0.f, 0.f, 0.f, 0.f, 0.f, 0.f, 0.f};
      s = WMMA_F16(qa0, load_bfrag(&Ks[nt * 16 + ml][0], hi), s);
      s = WMMA_F16(qa1, load_bfrag(&Ks[nt * 16 + ml][32], hi), s);
      sacc[nt] = s;
    }

    // ---- online softmax
#pragma unroll
    for (int r = 0; r < 8; r++) {
      int qrow = qw + hi * 8 + r;
      float sv[4];
      float mx = -__builtin_inff();
#pragma unroll
      for (int nt = 0; nt < 4; nt++) {
        int key = j0 + nt * 16 + ml;
        float v = sacc[nt][r];
        v = (key <= qrow) ? v : -__builtin_inff();
        sv[nt] = v;
        mx = fmaxf(mx, v);
      }
#pragma unroll
      for (int off = 1; off < 16; off <<= 1) mx = fmaxf(mx, __shfl_xor(mx, off, 32));
      float mo = m_run[r];
      float mn = fmaxf(mo, mx);
      float scale = (mo == mn) ? 1.0f : __expf(mo - mn);
      float psum = 0.0f;
#pragma unroll
      for (int nt = 0; nt < 4; nt++) {
        float p = (sv[nt] == -__builtin_inff()) ? 0.0f : __expf(sv[nt] - mn);
        psum += p;
        Pw[hi * 8 + r][nt * 16 + ml] = (half_t)p;
      }
#pragma unroll
      for (int off = 1; off < 16; off <<= 1) psum += __shfl_xor(psum, off, 32);
      m_run[r] = mn;
      l_run[r] = l_run[r] * scale + psum;
#pragma unroll
      for (int i = 0; i < 4; i++) o[i][r] *= scale;
    }

    // ---- O += P x V
    v16h pa0 = load_afrag(&Pw[ml][0], hi);
    v16h pa1 = load_afrag(&Pw[ml][32], hi);
#pragma unroll
    for (int i = 0; i < 4; i++) {
      o[i] = WMMA_F16(pa0, load_bfrag(&VT[i * 16 + ml][0], hi), o[i]);
      o[i] = WMMA_F16(pa1, load_bfrag(&VT[i * 16 + ml][32], hi), o[i]);
    }
  }

#pragma unroll
  for (int i = 0; i < 4; i++) {
#pragma unroll
    for (int r = 0; r < 8; r++) {
      int qrow = qw + hi * 8 + r;
      int col = h * HD + i * 16 + ml;
      outp[(size_t)(b * T + qrow) * C + col] = (half_t)(o[i][r] / l_run[r]);
    }
  }
}

// ---------------------------------------------------------------------------
// Launch
// ---------------------------------------------------------------------------
extern "C" void kernel_launch(void* const* d_in, const int* in_sizes, int n_in,
                              void* d_out, int out_size, void* d_ws, size_t ws_size,
                              hipStream_t stream) {
  (void)in_sizes; (void)n_in; (void)out_size; (void)ws_size;
  const float* x      = (const float*)d_in[0];
  const float* ln1_g  = (const float*)d_in[1];
  const float* ln1_b  = (const float*)d_in[2];
  const float* w_attn = (const float*)d_in[3];
  const float* b_attn = (const float*)d_in[4];
  const float* w_proj = (const float*)d_in[5];
  const float* b_proj = (const float*)d_in[6];
  const float* ln2_g  = (const float*)d_in[7];
  const float* ln2_b  = (const float*)d_in[8];
  const float* w_fc   = (const float*)d_in[9];
  const float* b_fc   = (const float*)d_in[10];
  const float* w_fc2  = (const float*)d_in[11];
  const float* b_fc2  = (const float*)d_in[12];
  float* out = (float*)d_out;

  char* ws = (char*)d_ws;
  size_t off = 0;
  auto take = [&](size_t bytes) -> char* {
    char* p = ws + off;
    off = (off + bytes + 255) & ~(size_t)255;
    return p;
  };
  // transposed f16 weights [N][K]
  half_t* wattnT = (half_t*)take((size_t)C3 * C * 2);
  half_t* wprojT = (half_t*)take((size_t)C * C * 2);
  half_t* wfcT   = (half_t*)take((size_t)4 * C * C * 2);
  half_t* wfc2T  = (half_t*)take((size_t)C * 4 * C * 2);
  half_t* h16    = (half_t*)take((size_t)MTOK * C * 2);
  half_t* qkv16  = (half_t*)take((size_t)MTOK * C3 * 2);
  half_t* attn16 = (half_t*)take((size_t)MTOK * C * 2);
  half_t* act16  = (half_t*)take((size_t)MTOK * 4 * C * 2);

  cvt_t_f16_k<<<dim3(C3 / 32, C / 32), 256, 0, stream>>>(w_attn, wattnT, C, C3);
  cvt_t_f16_k<<<dim3(C / 32, C / 32), 256, 0, stream>>>(w_proj, wprojT, C, C);
  cvt_t_f16_k<<<dim3(4 * C / 32, C / 32), 256, 0, stream>>>(w_fc, wfcT, C, 4 * C);
  cvt_t_f16_k<<<dim3(C / 32, 4 * C / 32), 256, 0, stream>>>(w_fc2, wfc2T, 4 * C, C);

  ln_f16_k<<<MTOK, 256, 0, stream>>>(x, ln1_g, ln1_b, h16);

  gemm_wmma_k<true, false, false><<<dim3(C3 / 64, MTOK / 128), 256, 0, stream>>>(
      h16, wattnT, b_attn, nullptr, qkv16, MTOK, C3, C);

  attn_wmma_k<<<dim3(T / 128, NB * H), 256, 0, stream>>>(qkv16, attn16);

  gemm_wmma_k<false, false, true><<<dim3(C / 64, MTOK / 128), 256, 0, stream>>>(
      attn16, wprojT, b_proj, x, (void*)out, MTOK, C, C);

  ln_f16_k<<<MTOK, 256, 0, stream>>>(out, ln2_g, ln2_b, h16);

  gemm_wmma_k<true, true, false><<<dim3(4 * C / 64, MTOK / 128), 256, 0, stream>>>(
      h16, wfcT, b_fc, nullptr, act16, MTOK, 4 * C, C);

  gemm_wmma_k<false, false, true><<<dim3(C / 64, MTOK / 128), 256, 0, stream>>>(
      act16, wfc2T, b_fc2, out, (void*)out, MTOK, C, 4 * C);
}
